// Gate_78168404787628
// MI455X (gfx1250) — matrix operand order repaired and checked by reading, not compiled, not run on hardware
//
#include <hip/hip_runtime.h>
#include <hip/hip_bf16.h>
#include <stdint.h>

// ---- problem constants (match reference) ----
#define BTOK   32768
#define DIM    4096
#define NEXP   256
#define NGRP   8
#define TOPG   4
#define TOPKE  8
#define RSCALE 2.5f

// ---- tiling ----
#define BM      64          // tokens per block
#define KCHUNK  64          // K elements staged per iteration
#define NCH     (DIM / KCHUNK)
#define ASTRIDE 72          // LDS row stride (shorts) for A tile, padded
#define WSTRIDE 72          // LDS row stride (shorts) for W tile, padded
#define SSTRIDE 257         // LDS row stride (floats) for score tile, padded

#define ABYTES  (BM * ASTRIDE * 2)          // 9216
#define WBYTES  (NEXP * WSTRIDE * 2)        // 36864
#define SMEM_BYTES (2 * ABYTES + 2 * WBYTES) // 92160; score tile (65792 B) overlays

typedef __attribute__((ext_vector_type(16))) __bf16 v16bf;
typedef __attribute__((ext_vector_type(8)))  __bf16 v8bf;
typedef __attribute__((ext_vector_type(8)))  float  v8f;

__device__ __forceinline__ unsigned short f2bf(float f) {
    union { float f; unsigned int u; } c; c.f = f;
    unsigned int u = c.u;
    u += 0x7FFFu + ((u >> 16) & 1u);   // round-to-nearest-even
    return (unsigned short)(u >> 16);
}

// Kernel 1: W f32 [E,DIM] -> bf16 in workspace (2 MB, stays L2 resident)
__global__ void __launch_bounds__(256)
wcvt_kernel(const float* __restrict__ W, unsigned short* __restrict__ Wb) {
    int i = (blockIdx.x * 256 + threadIdx.x) * 4;
    float4 f = *(const float4*)(W + i);
    union { unsigned short s[4]; uint2 v; } o;
    o.s[0] = f2bf(f.x); o.s[1] = f2bf(f.y);
    o.s[2] = f2bf(f.z); o.s[3] = f2bf(f.w);
    *(uint2*)(Wb + i) = o.v;
}

// ---- helpers for the GEMM kernel ----

// Issue 8 async global->LDS 16B copies of this thread's share of the W chunk.
// i = tid + j*256 indexes uint4s over the [256][KCHUNK] bf16 chunk.
__device__ __forceinline__ void issue_w_async(const unsigned short* __restrict__ Wg,
                                              unsigned short* Wdst, int k0, int tid) {
#pragma unroll
    for (int j = 0; j < 8; ++j) {
        int i    = tid + j * 256;
        int row  = i >> 3;
        int col  = (i & 7) * 8;
        unsigned goff = (unsigned)((row * DIM + k0 + col) * 2);          // byte offset
        unsigned lds  = (unsigned)(uintptr_t)(Wdst + row * WSTRIDE + col); // LDS byte addr
        asm volatile("global_load_async_to_lds_b128 %0, %1, %2"
                     :: "v"(lds), "v"(goff), "s"(Wg) : "memory");
    }
}

__device__ __forceinline__ void load_x(const float* __restrict__ xptr, int k0, float4 f[4]) {
#pragma unroll
    for (int j = 0; j < 4; ++j) f[j] = *(const float4*)(xptr + k0 + j * 4);
}

__device__ __forceinline__ void store_a(unsigned short* Adst, int xr, int xseg,
                                        const float4 f[4]) {
    union { unsigned short s[8]; uint4 v; } a0, a1;
    a0.s[0] = f2bf(f[0].x); a0.s[1] = f2bf(f[0].y); a0.s[2] = f2bf(f[0].z); a0.s[3] = f2bf(f[0].w);
    a0.s[4] = f2bf(f[1].x); a0.s[5] = f2bf(f[1].y); a0.s[6] = f2bf(f[1].z); a0.s[7] = f2bf(f[1].w);
    a1.s[0] = f2bf(f[2].x); a1.s[1] = f2bf(f[2].y); a1.s[2] = f2bf(f[2].z); a1.s[3] = f2bf(f[2].w);
    a1.s[4] = f2bf(f[3].x); a1.s[5] = f2bf(f[3].y); a1.s[6] = f2bf(f[3].z); a1.s[7] = f2bf(f[3].w);
    *(uint4*)(Adst + xr * ASTRIDE + xseg)     = a0.v;
    *(uint4*)(Adst + xr * ASTRIDE + xseg + 8) = a1.v;
}

// A fragment (16x32 bf16): lanes 0-15 hold K{kk+0..15} of row l16,
// lanes 16-31 hold K{kk+16..31}.
__device__ __forceinline__ v16bf ldfragA(const unsigned short* arow, int kk, int khalf) {
    v8bf lo = *(const v8bf*)(arow + kk + khalf * 16);
    v8bf hi = *(const v8bf*)(arow + kk + khalf * 16 + 8);
    return __builtin_shufflevector(lo, hi, 0,1,2,3,4,5,6,7,8,9,10,11,12,13,14,15);
}

// B fragment (32x16 bf16): lane n holds column (expert) nrow, K rows kk+khalf*16..+15.
__device__ __forceinline__ v16bf ldfragB(const unsigned short* Wl, int nrow, int kk, int khalf) {
    const unsigned short* p = Wl + nrow * WSTRIDE + kk + khalf * 16;
    v8bf lo = *(const v8bf*)p;
    v8bf hi = *(const v8bf*)(p + 8);
    return __builtin_shufflevector(lo, hi, 0,1,2,3,4,5,6,7,8,9,10,11,12,13,14,15);
}

// Kernel 2: fused scores = sigmoid(x @ W^T + b) + group-limited top-k routing.
// One block = 64 tokens x 256 experts. 8 waves, wave tile = 2 m-tiles x 4 n-tiles.
__global__ void __launch_bounds__(256)
gate_kernel(const float* __restrict__ x,
            const unsigned short* __restrict__ Wg,
            const float* __restrict__ bias,
            float* __restrict__ out) {
    extern __shared__ unsigned char smem[];
    unsigned short* A0 = (unsigned short*)smem;
    unsigned short* A1 = (unsigned short*)(smem + ABYTES);
    unsigned short* W0 = (unsigned short*)(smem + 2 * ABYTES);
    unsigned short* W1 = (unsigned short*)(smem + 2 * ABYTES + WBYTES);
    float*          Sl = (float*)smem;   // [BM][SSTRIDE], used after the K loop

    const int tid   = threadIdx.x;
    const int lane  = tid & 31;
    const int wave  = tid >> 5;
    const int mhalf = wave & 1;          // m-tiles {2*mhalf, 2*mhalf+1}
    const int nbase = (wave >> 1) * 64;  // 4 n-tiles of 16 experts
    const int l16   = lane & 15;
    const int khalf = lane >> 4;

    const long long rowBase = (long long)blockIdx.x * BM;

    v8f acc[8] = {v8f{}, v8f{}, v8f{}, v8f{}, v8f{}, v8f{}, v8f{}, v8f{}};

    // x staging: thread t -> row (t>>2), 16 consecutive floats at (t&3)*16
    const int xr   = tid >> 2;
    const int xseg = (tid & 3) * 16;
    const float* xptr = x + (rowBase + xr) * (long long)DIM + xseg;

    // ---- prologue: fill buffer 0 for chunk 0 ----
    float4 xf[4];
    load_x(xptr, 0, xf);
    issue_w_async(Wg, W0, 0, tid);
    store_a(A0, xr, xseg, xf);

    for (int kc = 0; kc < NCH; ++kc) {
        unsigned short* Acur = (kc & 1) ? A1 : A0;
        unsigned short* Wcur = (kc & 1) ? W1 : W0;
        unsigned short* Anxt = (kc & 1) ? A0 : A1;
        unsigned short* Wnxt = (kc & 1) ? W0 : W1;
        const bool more = (kc + 1) < NCH;

        if (more) load_x(xptr, (kc + 1) * KCHUNK, xf);   // global loads in flight early

        asm volatile("s_wait_asynccnt 0x0" ::: "memory"); // W(cur) async copies done
        __syncthreads();                                  // cur visible; nxt readers retired

        if (more) {
            issue_w_async(Wg, Wnxt, (kc + 1) * KCHUNK, tid); // overlaps with WMMAs below
            store_a(Anxt, xr, xseg, xf);
        }

#pragma unroll
        for (int ks = 0; ks < 2; ++ks) {
            const int kk = ks * 32;
            const unsigned short* ar0 = Acur + (mhalf * 32 + l16) * ASTRIDE;
            // distinct registers for every fragment -> all ds_loads issue before WMMAs
            v16bf a0 = ldfragA(ar0, kk, khalf);
            v16bf a1 = ldfragA(ar0 + 16 * ASTRIDE, kk, khalf);
            v16bf b0 = ldfragB(Wcur, nbase +  0 + l16, kk, khalf);
            v16bf b1 = ldfragB(Wcur, nbase + 16 + l16, kk, khalf);
            v16bf b2 = ldfragB(Wcur, nbase + 32 + l16, kk, khalf);
            v16bf b3 = ldfragB(Wcur, nbase + 48 + l16, kk, khalf);
            acc[0] = __builtin_amdgcn_wmma_f32_16x16x32_bf16(false, a0, false, b0, (short)0, acc[0], false, false);
            acc[1] = __builtin_amdgcn_wmma_f32_16x16x32_bf16(false, a0, false, b1, (short)0, acc[1], false, false);
            acc[2] = __builtin_amdgcn_wmma_f32_16x16x32_bf16(false, a0, false, b2, (short)0, acc[2], false, false);
            acc[3] = __builtin_amdgcn_wmma_f32_16x16x32_bf16(false, a0, false, b3, (short)0, acc[3], false, false);
            acc[4] = __builtin_amdgcn_wmma_f32_16x16x32_bf16(false, a1, false, b0, (short)0, acc[4], false, false);
            acc[5] = __builtin_amdgcn_wmma_f32_16x16x32_bf16(false, a1, false, b1, (short)0, acc[5], false, false);
            acc[6] = __builtin_amdgcn_wmma_f32_16x16x32_bf16(false, a1, false, b2, (short)0, acc[6], false, false);
            acc[7] = __builtin_amdgcn_wmma_f32_16x16x32_bf16(false, a1, false, b3, (short)0, acc[7], false, false);
        }
    }

    __syncthreads();   // all waves done with A/W buffers -> overlay the score tile

    // D layout: VGPR v, lanes 0-15: (M=v, N=lane); lanes 16-31: (M=v+8, N=lane-16)
#pragma unroll
    for (int mi = 0; mi < 2; ++mi)
#pragma unroll
        for (int j = 0; j < 4; ++j)
#pragma unroll
            for (int v = 0; v < 8; ++v) {
                int m = mhalf * 32 + mi * 16 + v + 8 * khalf;
                int n = nbase + j * 16 + l16;
                Sl[m * SSTRIDE + n] = acc[mi * 4 + j][v];
            }
    __syncthreads();

    // bias + sigmoid: thread t owns expert column t across the 64 token rows
    {
        float be = bias[tid];
        for (int r = 0; r < BM; ++r) {
            float s = Sl[r * SSTRIDE + tid];
            s = 1.0f / (1.0f + __expf(-(s + be)));
            Sl[r * SSTRIDE + tid] = s;
        }
    }
    __syncthreads();

    // routing: thread t<64 handles token rowBase+t
    if (tid < BM) {
        const float* row = Sl + tid * SSTRIDE;

        float gs[NGRP];
#pragma unroll
        for (int g = 0; g < NGRP; ++g) {
            float m1 = -1e30f, m2 = -1e30f;
            for (int e = g * 32; e < g * 32 + 32; ++e) {
                float s = row[e];
                if (s > m1) { m2 = m1; m1 = s; }
                else if (s > m2) { m2 = s; }
            }
            gs[g] = m1 + m2;
        }
        unsigned keep = 0;
#pragma unroll
        for (int rep = 0; rep < TOPG; ++rep) {
            float best = -1e30f; int bi = 0;
            for (int g = 0; g < NGRP; ++g)
                if (!((keep >> g) & 1) && gs[g] > best) { best = gs[g]; bi = g; }
            keep |= 1u << bi;
        }
        float tw[TOPKE]; int ti[TOPKE];
#pragma unroll
        for (int k = 0; k < TOPKE; ++k) { tw[k] = -1e30f; ti[k] = 0; }
        for (int g = 0; g < NGRP; ++g) {
            if (!((keep >> g) & 1)) continue;
            for (int e = g * 32; e < g * 32 + 32; ++e) {
                float s = row[e];
                if (s > tw[TOPKE - 1]) {
                    int p = TOPKE - 1;
                    while (p > 0 && s > tw[p - 1]) {
                        tw[p] = tw[p - 1]; ti[p] = ti[p - 1]; --p;
                    }
                    tw[p] = s; ti[p] = e;
                }
            }
        }
        float sum = 0.f;
#pragma unroll
        for (int k = 0; k < TOPKE; ++k) sum += tw[k];
        const float inv = RSCALE / sum;
        const long long token = rowBase + tid;
#pragma unroll
        for (int k = 0; k < TOPKE; ++k) {
            out[token * TOPKE + k] = tw[k] * inv;                            // weights
            out[(long long)BTOK * TOPKE + token * TOPKE + k] = (float)ti[k]; // indices
        }
    }
}

extern "C" void kernel_launch(void* const* d_in, const int* in_sizes, int n_in,
                              void* d_out, int out_size, void* d_ws, size_t ws_size,
                              hipStream_t stream) {
    (void)in_sizes; (void)n_in; (void)out_size; (void)ws_size;
    const float* x    = (const float*)d_in[0];   // [B, DIM] f32
    const float* W    = (const float*)d_in[1];   // [E, DIM] f32
    const float* bias = (const float*)d_in[2];   // [E] f32
    float* out = (float*)d_out;                  // [B*8 weights][B*8 indices]
    unsigned short* Wb = (unsigned short*)d_ws;  // bf16 W, 2 MB

    wcvt_kernel<<<(NEXP * DIM) / (256 * 4), 256, 0, stream>>>(W, Wb);
    gate_kernel<<<BTOK / BM, 256, SMEM_BYTES, stream>>>(x, Wb, bias, out);
}
